// ReccurentNN_28028956574207
// MI455X (gfx1250) — matrix operand order repaired and checked by reading, not compiled, not run on hardware
//
#include <hip/hip_runtime.h>

// ---------------- problem constants (match reference) ----------------
#define SEQ   2048
#define BATCH 128
#define EDIM  128
#define LDIM  128
#define G4    (4 * LDIM)   // 512 gate columns (f,i,g,o)
#define NC    2

typedef __attribute__((ext_vector_type(16))) _Float16 v16h;
typedef __attribute__((ext_vector_type(8)))  _Float16 v8h;
typedef __attribute__((ext_vector_type(8)))  float    v8f;

// global (addrspace(1)) pointer types: keep provenance through asm laundering
// so streamed weight loads stay global_load_b128 (LOADcnt only, partial waits)
typedef const __attribute__((address_space(1))) _Float16* gp16;
typedef const __attribute__((address_space(1))) v16h*     gpv16;

union AB16 {            // assemble a 16x32 f16 A-operand from two 16B chunks
    v16h v;
    v8h  h[2];
};

#define WMMA_F16(A, B, C) \
    __builtin_amdgcn_wmma_f32_16x16x32_f16(false, (A), false, (B), (short)0, (C), false, false)

// ---------------- weight prep: transpose + f16 hi/lo split ----------------
// WiT_hi/lo, WhT_hi/lo are [G4][K] row-major (one gate column per row) so a
// lane can load its B-operand as one contiguous 32-byte span.
__global__ void prep_kernel(const float* __restrict__ Wi, const float* __restrict__ bi,
                            const float* __restrict__ Wh, const float* __restrict__ bh,
                            _Float16* __restrict__ WiT_hi, _Float16* __restrict__ WiT_lo,
                            _Float16* __restrict__ WhT_hi, _Float16* __restrict__ WhT_lo,
                            float* __restrict__ bsum) {
    int idx = blockIdx.x * blockDim.x + threadIdx.x;   // 0 .. G4*128-1
    int g = idx >> 7;          // gate column 0..511
    int k = idx & 127;         // K index     0..127

    float wi = Wi[k * G4 + g];                 // Wi is [E][4L]
    _Float16 wih = (_Float16)wi;
    WiT_hi[g * EDIM + k] = wih;
    WiT_lo[g * EDIM + k] = (_Float16)(wi - (float)wih);

    float wh = Wh[k * G4 + g];                 // Wh is [L][4L]
    _Float16 whh = (_Float16)wh;
    WhT_hi[g * LDIM + k] = whh;
    WhT_lo[g * LDIM + k] = (_Float16)(wh - (float)whh);

    if (k == 0) bsum[g] = bi[g] + bh[g];
}

// ---------------- fused LSTM scan + classifier ----------------
__device__ __forceinline__ float fsigmoid(float v) {
    return 1.0f / (1.0f + __expf(-v));         // exp(+inf) -> denom inf -> 0: safe
}
__device__ __forceinline__ float ftanh(float v) {
    v = fminf(fmaxf(v, -15.0f), 15.0f);        // avoid inf/inf
    float e = __expf(-2.0f * v);
    return (1.0f - e) / (1.0f + e);
}

__launch_bounds__(256, 1)
__global__ void lstm_kernel(const float* __restrict__ x,
                            const _Float16* __restrict__ WiT_hi, const _Float16* __restrict__ WiT_lo,
                            const _Float16* __restrict__ WhT_hi, const _Float16* __restrict__ WhT_lo,
                            const float* __restrict__ bsum,
                            const float* __restrict__ Wc, const float* __restrict__ bc,
                            float* __restrict__ out) {
    // 32 KB static LDS: double-buffered h and x tiles, f16 hi/lo split.
    // Double buffering => a single barrier per timestep.
    __shared__ _Float16 sh_h_hi[2][16 * LDIM], sh_h_lo[2][16 * LDIM];
    __shared__ _Float16 sh_x_hi[2][16 * EDIM], sh_x_lo[2][16 * EDIM];

    const int tid  = threadIdx.x;
    const int lane = tid & 31;               // wave32 (gfx1250)
    const int wave = tid >> 5;               // 0..7 -> l-tile owned by this wave
    const int rb   = blockIdx.x * 16;        // batch-row base (independent recurrence)

    // x staging coordinates (whole block covers the 16x128 tile, 8 floats/thread)
    const int xrow = tid >> 4;               // 0..15
    const int xe0  = (tid & 15) * 8;         // 0..120

    // zero initial hidden state (buffer 0)
    for (int i = tid; i < 16 * LDIM; i += 256) {
        sh_h_hi[0][i] = (_Float16)0.0f;
        sh_h_lo[0][i] = (_Float16)0.0f;
    }

    // per-lane WMMA operand addressing (A and B both use contiguous k = 16*(lane/16) + s)
    const int col = lane & 15;               // tile column (B/C/D) == tile row (A)
    const int hk  = (lane >> 4) * 8;         // unused; A loads use bk (two 8-elem chunks at bk, bk+8)
    const int bk  = (lane >> 4) * 16;        // B-operand K sub-offset (one 16-elem span)

    const int lbase   = wave * 16;           // this wave's l-tile columns
    const int gcol[4] = { lbase, LDIM + lbase, 2 * LDIM + lbase, 3 * LDIM + lbase };

    float bias[4];
    #pragma unroll
    for (int g = 0; g < 4; ++g) bias[g] = bsum[gcol[g] + col];

    // Per-lane base pointers for streamed Wi: all 32 loads/step become
    // [VGPR base + constant imm offset] (g*32768 + kc*64 bytes).
    gp16 wb_hi = (gp16)(WiT_hi + (size_t)(lbase + col) * EDIM + bk);
    gp16 wb_lo = (gp16)(WiT_lo + (size_t)(lbase + col) * EDIM + bk);

    // Register-resident Wh hi+lo: 4 gate tiles x 4 K-chunks x 2 = 256 VGPRs.
    // These are reused 2048 times -> pinning them is the right trade.
    v16h whh[4][4], whl[4][4];
    #pragma unroll
    for (int g = 0; g < 4; ++g) {
        const _Float16* ph = WhT_hi + (size_t)(gcol[g] + col) * LDIM;
        const _Float16* pl = WhT_lo + (size_t)(gcol[g] + col) * LDIM;
        #pragma unroll
        for (int kc = 0; kc < 4; ++kc) {
            whh[g][kc] = *(const v16h*)(ph + kc * 32 + bk);
            whl[g][kc] = *(const v16h*)(pl + kc * 32 + bk);
        }
    }

    // persistent cell state: one 16x16 f32 accumulator tile per wave
    v8f cacc;
    #pragma unroll
    for (int j = 0; j < 8; ++j) cacc[j] = 0.0f;

    // stage x_0 into buffer 0
    {
        const float* px = x + ((size_t)0 * BATCH + rb + xrow) * EDIM + xe0;
        #pragma unroll
        for (int j = 0; j < 8; ++j) {
            float v = px[j];
            _Float16 hi = (_Float16)v;
            sh_x_hi[0][xrow * EDIM + xe0 + j] = hi;
            sh_x_lo[0][xrow * EDIM + xe0 + j] = (_Float16)(v - (float)hi);
        }
    }
    __syncthreads();

    for (int t = 0; t < SEQ; ++t) {
        const int cur = t & 1;
        const int nxt = cur ^ 1;

        // Launder the streamed-weight base pointers (address_space(1) is kept,
        // so loads remain global_load_b128). This makes them formally
        // loop-variant: LICM cannot hoist the 32 weight loads into registers
        // and force spills; they stay in-loop L2 hits (weights are 0.5 MB,
        // permanently resident in the 192 MB L2).
        gp16 p_hi = wb_hi, p_lo = wb_lo;
        asm volatile("" : "+v"(p_hi), "+v"(p_lo));

        // ---- (1) issue x_{t+1} global loads now; WMMA phase hides the latency
        float xr[8];
        {
            const int tn = (t + 1 < SEQ) ? (t + 1) : t;   // uniform; last iter redundant
            const float* px = x + ((size_t)tn * BATCH + rb + xrow) * EDIM + xe0;
            #pragma unroll
            for (int j = 0; j < 8; ++j) xr[j] = px[j];
            __builtin_prefetch(px + BATCH * EDIM, 0, 1);  // warm t+2 (global_prefetch_b8)
        }

        // ---- (2) gate accumulators seeded with zero (inline SRC2=0 on 1st WMMA);
        //          bi+bh is folded into the pointwise phase instead.
        v8f acc[4];
        #pragma unroll
        for (int g = 0; g < 4; ++g)
            #pragma unroll
            for (int j = 0; j < 8; ++j) acc[g][j] = 0.0f;

        // ---- x_t @ Wi : B streamed from L2-resident split weights ----
        #pragma unroll
        for (int kc = 0; kc < 4; ++kc) {
            AB16 ah, al;
            const _Float16* axh = &sh_x_hi[cur][col * EDIM + kc * 32];
            const _Float16* axl = &sh_x_lo[cur][col * EDIM + kc * 32];
            ah.h[0] = *(const v8h*)(axh + bk);
            ah.h[1] = *(const v8h*)(axh + bk + 8);
            al.h[0] = *(const v8h*)(axl + bk);
            al.h[1] = *(const v8h*)(axl + bk + 8);
            #pragma unroll
            for (int g = 0; g < 4; ++g) {
                v16h bh = *(gpv16)(p_hi + g * (LDIM * EDIM) + kc * 32);
                v16h bl = *(gpv16)(p_lo + g * (LDIM * EDIM) + kc * 32);
                acc[g] = WMMA_F16(ah.v, bh, acc[g]);   // hi*hi
                acc[g] = WMMA_F16(ah.v, bl, acc[g]);   // hi*lo
                acc[g] = WMMA_F16(al.v, bh, acc[g]);   // lo*hi
            }
        }

        // ---- h_{t-1} @ Wh : B operands register-resident ----
        #pragma unroll
        for (int kc = 0; kc < 4; ++kc) {
            AB16 ah, al;
            const _Float16* axh = &sh_h_hi[cur][col * LDIM + kc * 32];
            const _Float16* axl = &sh_h_lo[cur][col * LDIM + kc * 32];
            ah.h[0] = *(const v8h*)(axh + bk);
            ah.h[1] = *(const v8h*)(axh + bk + 8);
            al.h[0] = *(const v8h*)(axl + bk);
            al.h[1] = *(const v8h*)(axl + bk + 8);
            #pragma unroll
            for (int g = 0; g < 4; ++g) {
                acc[g] = WMMA_F16(ah.v, whh[g][kc], acc[g]);
                acc[g] = WMMA_F16(ah.v, whl[g][kc], acc[g]);
                acc[g] = WMMA_F16(al.v, whh[g][kc], acc[g]);
            }
        }

        // ---- (3) park x_{t+1} into the other LDS buffer (loads now complete)
        #pragma unroll
        for (int j = 0; j < 8; ++j) {
            _Float16 hi = (_Float16)xr[j];
            sh_x_hi[nxt][xrow * EDIM + xe0 + j] = hi;
            sh_x_lo[nxt][xrow * EDIM + xe0 + j] = (_Float16)(xr[j] - (float)hi);
        }

        // ---- (4) pointwise LSTM cell update (C/D layout: VGPR j -> row j or 8+j)
        // ---- (5) publish h_t into the other buffer
        {
            const int hrow = (lane >> 4) * 8;
            #pragma unroll
            for (int j = 0; j < 8; ++j) {
                float f  = fsigmoid(acc[0][j] + bias[0]);
                float ig = fsigmoid(acc[1][j] + bias[1]);
                float gg = ftanh(acc[2][j] + bias[2]);
                float o  = fsigmoid(acc[3][j] + bias[3]);
                float c  = f * cacc[j] + ig * gg;
                cacc[j]  = c;
                float hv = o * ftanh(c);
                _Float16 hi = (_Float16)hv;
                int idx = (hrow + j) * LDIM + lbase + col;
                sh_h_hi[nxt][idx] = hi;
                sh_h_lo[nxt][idx] = (_Float16)(hv - (float)hi);
            }
        }

        // ---- (6) single barrier per step: next-step buffers are now valid
        __syncthreads();
    }

    // ---- tiny classifier: out[rb+r][c] = h[r,:] . Wc[:,c] + bc[c] ----
    // final h lives in buffer (SEQ & 1)
    if (tid < 32) {
        const int fb = SEQ & 1;
        int r = tid >> 1, cidx = tid & 1;
        float s = bc[cidx];
        for (int l = 0; l < LDIM; ++l) {
            float hv = (float)sh_h_hi[fb][r * LDIM + l] + (float)sh_h_lo[fb][r * LDIM + l];
            s += hv * Wc[l * NC + cidx];
        }
        out[(rb + r) * NC + cidx] = s;
    }
}

// ---------------- host entry ----------------
extern "C" void kernel_launch(void* const* d_in, const int* in_sizes, int n_in,
                              void* d_out, int out_size, void* d_ws, size_t ws_size,
                              hipStream_t stream) {
    const float* x  = (const float*)d_in[0];
    const float* Wi = (const float*)d_in[1];
    const float* bi = (const float*)d_in[2];
    const float* Wh = (const float*)d_in[3];
    const float* bh = (const float*)d_in[4];
    const float* Wc = (const float*)d_in[5];
    const float* bc = (const float*)d_in[6];
    float* out = (float*)d_out;

    // workspace layout (all 256B-aligned): 4 x 128KB split weights + bias sum
    char* ws = (char*)d_ws;
    _Float16* WiT_hi = (_Float16*)(ws);
    _Float16* WiT_lo = (_Float16*)(ws + 131072);
    _Float16* WhT_hi = (_Float16*)(ws + 262144);
    _Float16* WhT_lo = (_Float16*)(ws + 393216);
    float*    bsum   = (float*)   (ws + 524288);

    prep_kernel<<<(G4 * 128) / 256, 256, 0, stream>>>(Wi, bi, Wh, bh,
                                                      WiT_hi, WiT_lo, WhT_hi, WhT_lo, bsum);
    lstm_kernel<<<BATCH / 16, 256, 0, stream>>>(x, WiT_hi, WiT_lo, WhT_hi, WhT_lo,
                                                bsum, Wc, bc, out);
}
